// SecondOrderDeformableAlignment_32427003084993
// MI455X (gfx1250) — matrix operand order repaired and checked by reading, not compiled, not run on hardware
//
#include <hip/hip_runtime.h>
#include <hip/hip_bf16.h>

typedef __attribute__((ext_vector_type(16))) __bf16 v16bf;
typedef __attribute__((ext_vector_type(8)))  float  v8f;

#define T_D 5
#define H_D 64
#define W_D 64
#define SP  (T_D*H_D*W_D)      /* 20480 */
#define CIN 128
#define FIN 198
#define FPAD 224               /* 198 padded to 7*32 */
#define C3  324                /* 81*DG */
#define NT3 21                 /* ceil(324/16) */
#define CG  32                 /* channels per deform group */
#define MAXMAG 10.0f
#define NEGS   0.1f

// ---- gfx1250 async copy global->LDS (ASYNCcnt path), per-lane 16B
__device__ __forceinline__ void async_b128(unsigned lds_addr, const __bf16* g) {
  asm volatile("global_load_async_to_lds_b128 %0, %1, off"
               :: "v"(lds_addr), "v"(g) : "memory");
}
__device__ __forceinline__ void wait_async0() {
  asm volatile("s_wait_asynccnt 0x0" ::: "memory");
}

// ---------------- feat packing: concat(extra, flow1, flow2) -> bf16 [S][224]
__global__ void k_pack_feat(const float* __restrict__ ef, const float* __restrict__ f1,
                            const float* __restrict__ f2, __bf16* __restrict__ out) {
  int idx = blockIdx.x * 256 + threadIdx.x;
  if (idx >= SP * FPAD) return;
  int p = idx / FPAD, c = idx - p * FPAD;
  float v = 0.f;
  if (c < 192)      v = ef[(size_t)c * SP + p];
  else if (c < 195) v = f1[(size_t)(c - 192) * SP + p];
  else if (c < 198) v = f2[(size_t)(c - 195) * SP + p];
  out[(size_t)p * FPAD + c] = (__bf16)v;
}

// ---------------- stage conv weights into per-lane WMMA B-fragment order
// B layout: lane half h, elem e holds K = 8h + (e&7) + 16*(e>>3).
__global__ void k_stage_w(const float* __restrict__ w, __bf16* __restrict__ out,
                          int cin, int kbN, int cout, int nt) {
  int total = 27 * kbN * nt * 512;
  int idx = blockIdx.x * 256 + threadIdx.x;
  if (idx >= total) return;
  int e    = idx & 15;
  int lane = (idx >> 4) & 31;
  int frag = idx >> 9;
  int ntile = frag % nt;
  int kb    = (frag / nt) % kbN;
  int tap   = frag / (nt * kbN);
  int n = ntile * 16 + (lane & 15);
  int k = kb * 32 + (lane >> 4) * 8 + (e & 7) + ((e >> 3) << 4);
  float v = 0.f;
  if (n < cout && k < cin) v = w[((size_t)n * cin + k) * 27 + tap];
  out[idx] = (__bf16)v;
}

// ---------------- stage DCN weights: frag index = (g*27+k27)*4 + ntile
__global__ void k_stage_wdcn(const float* __restrict__ w, __bf16* __restrict__ out) {
  int idx = blockIdx.x * 256 + threadIdx.x;
  if (idx >= 108 * 4 * 512) return;
  int e    = idx & 15;
  int lane = (idx >> 4) & 31;
  int frag = idx >> 9;
  int ntile = frag & 3;
  int gk    = frag >> 2;
  int g = gk / 27, k27 = gk % 27;
  int n = ntile * 16 + (lane & 15);
  int c = (lane >> 4) * 8 + (e & 7) + ((e >> 3) << 4);          // 0..31 within group
  out[idx] = (__bf16)w[((size_t)n * CIN + g * CG + c) * 27 + k27];
}

// ---------------- implicit-GEMM 3x3x3 conv via WMMA bf16 -> f32
// block = 128 threads (4 waves). M-tile = 64 spatial (one full W row).
// For each (kd,kh): async-stage the input row (64 x CINPAD, contiguous) into
// LDS once per block, then all 4 waves build A-fragments from LDS for the 3 kw
// shifts. Wave handles one N-tile of 16 channels and all 4 M-subtiles.
template<int CINPAD, int COUTR, int NTTOT, int ACT, int OUTF32>
__global__ void k_conv(const __bf16* __restrict__ in, const __bf16* __restrict__ wB,
                       const float* __restrict__ bias, void* __restrict__ outp) {
  constexpr int KB = CINPAD / 32;
  constexpr int CHUNKS = (64 * CINPAD) / (128 * 8);   // b128 copies per thread
  __shared__ __bf16 tileA[64 * CINPAD];
  int tid  = threadIdx.x;
  int lane = tid & 31;
  int wave = tid >> 5;
  int ntile = blockIdx.y * 4 + wave;
  bool active = (ntile < NTTOT);       // wave-uniform; all waves still barrier
  int mBase = blockIdx.x * 64;
  int t = mBase >> 12;
  int h = (mBase >> 6) & 63;
  int hi = lane >> 4;
  int lo = lane & 15;
  unsigned ldsBase = (unsigned)(uintptr_t)(&tileA[0]);

  v8f acc[4] = {};
  for (int dk = 0; dk < 3; ++dk) {
    int tt = t + dk - 1;
    if (tt < 0 || tt >= T_D) continue;            // block-uniform
    for (int hk = 0; hk < 3; ++hk) {
      int th = h + hk - 1;
      if (th < 0 || th >= H_D) continue;          // block-uniform
      const __bf16* rowBase = in + (size_t)((tt * H_D + th) * W_D) * CINPAD;
#pragma unroll
      for (int i = 0; i < CHUNKS; ++i) {
        int elt = (tid + i * 128) * 8;            // 8 bf16 = 16B per copy
        async_b128(ldsBase + (unsigned)(elt * 2), rowBase + elt);
      }
      wait_async0();
      __syncthreads();
      if (active) {
#pragma unroll
        for (int wk = 0; wk < 3; ++wk) {
          int tap = (dk * 3 + hk) * 3 + wk;
#pragma unroll
          for (int kb = 0; kb < KB; ++kb) {
            v16bf bfr = *(const v16bf*)(wB + ((size_t)(tap * KB + kb) * NTTOT + ntile) * 512 + lane * 16);
#pragma unroll
            for (int ms = 0; ms < 4; ++ms) {
              int ww = ms * 16 + lo + wk - 1;
              union { v16bf v; float4 f4[2]; } a;
              if ((unsigned)ww < (unsigned)W_D) {
                const __bf16* src = &tileA[ww * CINPAD + kb * 32 + hi * 8];
                a.f4[0] = *(const float4*)(src);        // k-local {0..7}+hi*8
                a.f4[1] = *(const float4*)(src + 16);   // k-local {16..23}+hi*8
              } else {
                a.f4[0] = make_float4(0.f, 0.f, 0.f, 0.f);
                a.f4[1] = make_float4(0.f, 0.f, 0.f, 0.f);
              }
              acc[ms] = __builtin_amdgcn_wmma_f32_16x16x32_bf16(
                  false, a.v, false, bfr, (short)0, acc[ms], false, false);
            }
          }
        }
      }
      __syncthreads();                            // protect tileA before restage
    }
  }
  int n = ntile * 16 + lo;
  if (active && n < COUTR) {
    float bv = bias[n];
#pragma unroll
    for (int ms = 0; ms < 4; ++ms) {
#pragma unroll
      for (int r = 0; r < 8; ++r) {
        int p = mBase + ms * 16 + hi * 8 + r;
        float v = acc[ms][r] + bv;
        if (ACT) v = (v >= 0.f) ? v : NEGS * v;
        if (OUTF32) ((float*)outp)[(size_t)p * COUTR + n] = v;
        else        ((__bf16*)outp)[(size_t)p * COUTR + n] = (__bf16)v;
      }
    }
  }
}

// ---------------- offset = 10*tanh(conv3) + flipped/tiled flow (in place)
__global__ void k_offset(float* __restrict__ o3, const float* __restrict__ f1,
                         const float* __restrict__ f2) {
  int idx = blockIdx.x * 256 + threadIdx.x;
  if (idx >= SP * C3) return;
  int p = idx / C3, ch = idx - p * C3;
  int comp = ch % 3;
  const float* fl = (ch < 162) ? f1 : f2;
  o3[idx] = MAXMAG * tanhf(o3[idx]) + fl[(size_t)(2 - comp) * SP + p];
}

// ---------------- deformable sampling + DCN reduction via WMMA
// block = 128 threads (4 waves), M-tile = 16 spatial, N = 64 outs (1 N-tile/wave).
// K loop: 108 chunks of 32 (g,k27 pairs x group channels).
__global__ void k_deform(const float* __restrict__ x, const float* __restrict__ offs,
                         const __bf16* __restrict__ wB, const float* __restrict__ bias,
                         float* __restrict__ out) {
  __shared__ __bf16 aTile[512];          // 16x32 bf16 A-tile in fragment order
  int tid  = threadIdx.x;
  int lane = tid & 31;
  int wave = tid >> 5;
  int pl   = tid & 15;                   // spatial row within tile
  int cg4  = tid >> 4;                   // 0..7 -> channel quad
  int pBase = blockIdx.x * 16;
  int t  = pBase >> 12;
  int h  = (pBase >> 6) & 63;
  int wb = pBase & 63;
  int p  = pBase + pl;

  v8f acc = {};
  for (int gk = 0; gk < 108; ++gk) {
    int g = gk / 27, k27 = gk % 27;
    int kd = k27 / 9 - 1, kh = (k27 / 3) % 3 - 1, kw = k27 % 3 - 1;
    const float* orow = offs + (size_t)p * C3 + (g * 27 + k27) * 3;
    float pd = orow[0] + (float)(t + kd);
    float ph = orow[1] + (float)(h + kh);
    float pw = orow[2] + (float)(wb + pl + kw);
    float d0 = floorf(pd), h0 = floorf(ph), w0 = floorf(pw);
    float fd = pd - d0, fh = ph - h0, fw = pw - w0;
    int lin[8]; float wv[8];
#pragma unroll
    for (int q = 0; q < 8; ++q) {
      int i = q >> 2, j = (q >> 1) & 1, l = q & 1;
      float dd = d0 + i, hh = h0 + j, wp = w0 + l;
      float wgt = (i ? fd : 1.f - fd) * (j ? fh : 1.f - fh) * (l ? fw : 1.f - fw);
      bool ok = (dd >= 0.f) && (dd <= (float)(T_D - 1)) &&
                (hh >= 0.f) && (hh <= (float)(H_D - 1)) &&
                (wp >= 0.f) && (wp <= (float)(W_D - 1));
      int di = (int)fminf(fmaxf(dd, 0.f), (float)(T_D - 1));
      int hj = (int)fminf(fmaxf(hh, 0.f), (float)(H_D - 1));
      int wl = (int)fminf(fmaxf(wp, 0.f), (float)(W_D - 1));
      lin[q] = (di * H_D + hj) * W_D + wl;
      wv[q]  = ok ? wgt : 0.f;
    }
#pragma unroll
    for (int cc = 0; cc < 4; ++cc) {
      int cl = cg4 * 4 + cc;             // 0..31 local K
      const float* xc = x + (size_t)(g * CG + cl) * SP;
      float s = 0.f;
#pragma unroll
      for (int q = 0; q < 8; ++q) s += wv[q] * xc[lin[q]];
      // inverse of A-fragment mapping: lane = m | (((k>>3)&1)<<4), e = (k&7)|((k&16)>>1)
      int laneT = pl + (((cl >> 3) & 1) << 4);
      int e = (cl & 7) | ((cl & 16) >> 1);
      aTile[laneT * 16 + e] = (__bf16)s;
    }
    __syncthreads();
    v16bf a = *(const v16bf*)(&aTile[lane * 16]);
    v16bf b = *(const v16bf*)(wB + ((size_t)(gk * 4) + wave) * 512 + lane * 16);
    acc = __builtin_amdgcn_wmma_f32_16x16x32_bf16(false, a, false, b, (short)0, acc, false, false);
    __syncthreads();
  }
  int n = wave * 16 + (lane & 15);
  float bv = bias[n];
  int hi = lane >> 4;
#pragma unroll
  for (int r = 0; r < 8; ++r) {
    out[(size_t)n * SP + pBase + hi * 8 + r] = acc[r] + bv;
  }
}

extern "C" void kernel_launch(void* const* d_in, const int* in_sizes, int n_in,
                              void* d_out, int out_size, void* d_ws, size_t ws_size,
                              hipStream_t stream) {
  const float* x  = (const float*)d_in[0];
  const float* ef = (const float*)d_in[1];
  const float* f1 = (const float*)d_in[2];
  const float* f2 = (const float*)d_in[3];
  const float* w1 = (const float*)d_in[4];
  const float* b1 = (const float*)d_in[5];
  const float* w2 = (const float*)d_in[6];
  const float* b2 = (const float*)d_in[7];
  const float* w3 = (const float*)d_in[8];
  const float* b3 = (const float*)d_in[9];
  const float* wd = (const float*)d_in[10];
  const float* bd = (const float*)d_in[11];
  float* out = (float*)d_out;

  char* ws = (char*)d_ws;
  size_t off = 0;
  auto alloc = [&](size_t bytes) -> void* {
    void* pp = ws + off;
    off = (off + bytes + 255) & ~(size_t)255;
    return pp;
  };
  __bf16* featPack = (__bf16*)alloc((size_t)SP * FPAD * 2);
  __bf16* act1     = (__bf16*)alloc((size_t)SP * 64 * 2);
  __bf16* act2     = (__bf16*)alloc((size_t)SP * 64 * 2);
  float*  o3       = (float*)alloc((size_t)SP * C3 * 4);
  __bf16* wB1      = (__bf16*)alloc((size_t)27 * 7 * 4 * 512 * 2);
  __bf16* wB2      = (__bf16*)alloc((size_t)27 * 2 * 4 * 512 * 2);
  __bf16* wB3      = (__bf16*)alloc((size_t)27 * 2 * NT3 * 512 * 2);
  __bf16* wBd      = (__bf16*)alloc((size_t)108 * 4 * 512 * 2);

  int n1 = SP * FPAD;
  k_pack_feat<<<(n1 + 255) / 256, 256, 0, stream>>>(ef, f1, f2, featPack);

  int nw1 = 27 * 7 * 4 * 512;
  k_stage_w<<<(nw1 + 255) / 256, 256, 0, stream>>>(w1, wB1, FIN, 7, 64, 4);
  int nw2 = 27 * 2 * 4 * 512;
  k_stage_w<<<(nw2 + 255) / 256, 256, 0, stream>>>(w2, wB2, 64, 2, 64, 4);
  int nw3 = 27 * 2 * NT3 * 512;
  k_stage_w<<<(nw3 + 255) / 256, 256, 0, stream>>>(w3, wB3, 64, 2, C3, NT3);
  int nwd = 108 * 4 * 512;
  k_stage_wdcn<<<(nwd + 255) / 256, 256, 0, stream>>>(wd, wBd);

  k_conv<FPAD, 64, 4, 1, 0><<<dim3(SP / 64, 1), 128, 0, stream>>>(featPack, wB1, b1, (void*)act1);
  k_conv<64,   64, 4, 1, 0><<<dim3(SP / 64, 1), 128, 0, stream>>>(act1, wB2, b2, (void*)act2);
  k_conv<64,  C3, NT3, 0, 1><<<dim3(SP / 64, 6), 128, 0, stream>>>(act2, wB3, b3, (void*)o3);

  int no = SP * C3;
  k_offset<<<(no + 255) / 256, 256, 0, stream>>>(o3, f1, f2);

  k_deform<<<SP / 16, 128, 0, stream>>>(x, o3, wBd, bd, out);
}
